// MultiViewGAT_63093069578891
// MI455X (gfx1250) — matrix operand order repaired and checked by reading, not compiled, not run on hardware
//
#include <hip/hip_runtime.h>
#include <cstdint>
#include <cstddef>

#define N_NODES 50000
#define N_EDGES 800000
#define HEADS   8
#define CH      16
#define F_IN    128
#define HC      128      // HEADS*CH
#define GFEAT   32
#define NBATCH  64
#define NCLS    10
#define SLOPE   0.2f
#define COMB    416      // 3*HC + GFEAT
#define M_TILES 3125     // N_NODES / 16
#define EP      (N_EDGES + N_NODES)  // edges + self loops

typedef __attribute__((ext_vector_type(16))) __bf16 v16bf;
typedef __attribute__((ext_vector_type(8)))  float  v8f;

union FragU { uint4 q[2]; v16bf v; };

__device__ __forceinline__ unsigned short f2bf(float f) {
  unsigned int u = __float_as_uint(f);
  u = u + 0x7FFFu + ((u >> 16) & 1u);   // round-to-nearest-even
  return (unsigned short)(u >> 16);
}

__device__ __forceinline__ float leaky(float x) { return x > 0.f ? x : SLOPE * x; }

// float atomic max supporting mixed signs (location initialized to -inf bits)
__device__ __forceinline__ void atomicMaxF(float* a, float v) {
  if (v >= 0.f) atomicMax((int*)a, __float_as_int(v));
  else          atomicMin((unsigned int*)a, __float_as_uint(v));
}

// ---------------- zero init (pooled sums + batch counts) ----------------
__global__ void k_zero(unsigned int* p, int nWords) {
  int i = blockIdx.x * blockDim.x + threadIdx.x;
  if (i < nWords) p[i] = 0u;
}

// ---------------- batch histogram (once) ----------------
__global__ void k_count_batch(const int* __restrict__ batch, int* __restrict__ cnt) {
  int i = blockIdx.x * blockDim.x + threadIdx.x;
  if (i < N_NODES) atomicAdd(&cnt[batch[i]], 1);
}

// ---------------- pack W [128][128] f32 -> bf16 WMMA B fragments ----------------
// layout: wp[kt(4)][nt(8)][lane(32)][e(16)];  B elem(lane,e) = W[kt*32 + 16*(lane>>4) + e][nt*16 + (lane&15)]
__global__ void k_pack_w(const float* __restrict__ W, unsigned short* __restrict__ wp) {
  int t = threadIdx.x;                 // 0..1023 = (kt,nt,lane)
  if (t >= 4 * 8 * 32) return;
  int lane = t & 31, nt = (t >> 5) & 7, kt = t >> 8;
  int hi = lane >> 4, col = lane & 15;
  unsigned short* dst = wp + (size_t)t * 16;
#pragma unroll
  for (int e = 0; e < 16; ++e) {
    int k = kt * 32 + 16 * hi + e;
    int n = nt * 16 + col;
    dst[e] = f2bf(W[k * HC + n]);
  }
}

// ---------------- x f32 -> bf16 (row-major [N][128]) ----------------
__global__ void k_cvt_bf16(const float* __restrict__ x, unsigned short* __restrict__ xb, int n) {
  int i = (blockIdx.x * blockDim.x + threadIdx.x) * 4;
  if (i >= n) return;
  float4 f = *(const float4*)(x + i);
  ushort4 o; o.x = f2bf(f.x); o.y = f2bf(f.y); o.z = f2bf(f.z); o.w = f2bf(f.w);
  *(ushort4*)(xb + i) = o;
}

// ---------------- h = x @ W via v_wmma_f32_16x16x32_bf16 ----------------
// one wave per 16-row M tile; 8 N-tiles (N=128) accumulated; K=128 in 4 steps of 32
__global__ void k_gemm(const unsigned short* __restrict__ xb,  // [N][128] bf16
                       const unsigned short* __restrict__ wp,  // [4][8][32][16] bf16
                       float* __restrict__ h) {                // [N][128] f32
  int wave = blockIdx.x * (blockDim.x >> 5) + (threadIdx.x >> 5);
  if (wave >= M_TILES) return;       // wave-uniform: EXEC stays all-ones for WMMA
  int lane = threadIdx.x & 31;
  int hi = lane >> 4, col = lane & 15;
  const unsigned short* xrow = xb + (size_t)(wave * 16 + col) * F_IN;

  v8f acc[8] = {};
#pragma unroll
  for (int kt = 0; kt < 4; ++kt) {
    int k0 = kt * 32;
    FragU a;
    a.q[0] = *(const uint4*)(xrow + k0 + 16 * hi);       // elems 0..7 : K = k0 + 16*hi + e
    a.q[1] = *(const uint4*)(xrow + k0 + 16 * hi + 8);   // elems 8..15: K = k0 + 16*hi + e
#pragma unroll
    for (int nt = 0; nt < 8; ++nt) {
      FragU b;
      const uint4* bp = (const uint4*)(wp + (size_t)(((kt * 8 + nt) * 32) + lane) * 16);
      b.q[0] = bp[0]; b.q[1] = bp[1];
      acc[nt] = __builtin_amdgcn_wmma_f32_16x16x32_bf16(
          false, a.v, false, b.v, (short)0, acc[nt], false, false);
    }
  }
  // C/D layout: lane l, vgpr j -> row = 16*tile + 8*(l>>4) + j, col = nt*16 + (l&15)
#pragma unroll
  for (int nt = 0; nt < 8; ++nt)
#pragma unroll
    for (int j = 0; j < 8; ++j)
      h[(size_t)(wave * 16 + 8 * hi + j) * HC + nt * 16 + col] = acc[nt][j];
}

// ---------------- attention dots + per-view state init ----------------
__global__ void k_attn_init(const float* __restrict__ h,
                            const float* __restrict__ att_src,   // [8][16]
                            const float* __restrict__ att_dst,
                            float* __restrict__ a_src, float* __restrict__ a_dst,
                            float* __restrict__ m, float* __restrict__ den,
                            float* __restrict__ outv) {
  int idx = blockIdx.x * blockDim.x + threadIdx.x;      // (n, head)
  if (idx >= N_NODES * HEADS) return;
  int n = idx >> 3, hd = idx & 7;
  const float* hp = h + (size_t)n * HC + hd * CH;
  float s = 0.f, d = 0.f;
#pragma unroll
  for (int c = 0; c < CH; ++c) {
    float v = hp[c];
    s += v * att_src[hd * CH + c];
    d += v * att_dst[hd * CH + c];
  }
  a_src[idx] = s;
  a_dst[idx] = d;
  m[idx] = -__builtin_inff();
  den[idx] = 0.f;
  float* op = outv + (size_t)n * HC + hd * CH;
#pragma unroll
  for (int c = 0; c < CH; ++c) op[c] = 0.f;
}

// ---------------- pass 1: segment max over incoming edges ----------------
__global__ void k_edge_max(const int* __restrict__ ei,
                           const float* __restrict__ a_src, const float* __restrict__ a_dst,
                           float* __restrict__ m) {
  int i = blockIdx.x * blockDim.x + threadIdx.x;
  if (i >= EP) return;
  int src = (i < N_EDGES) ? ei[i]            : (i - N_EDGES);
  int dst = (i < N_EDGES) ? ei[N_EDGES + i]  : (i - N_EDGES);
#pragma unroll
  for (int hd = 0; hd < HEADS; ++hd) {
    float l = leaky(a_src[src * HEADS + hd] + a_dst[dst * HEADS + hd]);
    atomicMaxF(&m[dst * HEADS + hd], l);
  }
}

// ---------------- pass 2: softmax denominator ----------------
__global__ void k_edge_sum(const int* __restrict__ ei,
                           const float* __restrict__ a_src, const float* __restrict__ a_dst,
                           const float* __restrict__ m, float* __restrict__ den) {
  int i = blockIdx.x * blockDim.x + threadIdx.x;
  if (i >= EP) return;
  int src = (i < N_EDGES) ? ei[i]            : (i - N_EDGES);
  int dst = (i < N_EDGES) ? ei[N_EDGES + i]  : (i - N_EDGES);
#pragma unroll
  for (int hd = 0; hd < HEADS; ++hd) {
    float l = leaky(a_src[src * HEADS + hd] + a_dst[dst * HEADS + hd]);
    atomicAdd(&den[dst * HEADS + hd], __expf(l - m[dst * HEADS + hd]));
  }
}

// ---------------- pass 3: weighted aggregate (1 wave per edge, 4 cols/lane) ----------------
__global__ void k_edge_aggr(const int* __restrict__ ei,
                            const float* __restrict__ a_src, const float* __restrict__ a_dst,
                            const float* __restrict__ m, const float* __restrict__ den,
                            const float* __restrict__ h, float* __restrict__ outv) {
  int wv = blockIdx.x * (blockDim.x >> 5) + (threadIdx.x >> 5);
  if (wv >= EP) return;
  int lane = threadIdx.x & 31;
  int src = (wv < N_EDGES) ? ei[wv]            : (wv - N_EDGES);
  int dst = (wv < N_EDGES) ? ei[N_EDGES + wv]  : (wv - N_EDGES);
  int hd = lane >> 2;                              // 4 consecutive cols share one head
  float l = leaky(a_src[src * HEADS + hd] + a_dst[dst * HEADS + hd]);
  float alpha = __expf(l - m[dst * HEADS + hd]) / den[dst * HEADS + hd];
  const float4 hv = *(const float4*)(h + (size_t)src * HC + lane * 4);
  float* op = outv + (size_t)dst * HC + lane * 4;
  atomicAdd(op + 0, alpha * hv.x);
  atomicAdd(op + 1, alpha * hv.y);
  atomicAdd(op + 2, alpha * hv.z);
  atomicAdd(op + 3, alpha * hv.w);
}

// ---------------- mean-pool partials (batch is sorted; flush on transition) ----------------
__global__ void k_pool(const float* __restrict__ outv, const int* __restrict__ batch,
                       float* __restrict__ pooled /* [64][128] for this view */) {
  int t = threadIdx.x;                 // column 0..127
  int n0 = blockIdx.x * 256;
  float acc = 0.f; int curb = -1;
  for (int i = 0; i < 256; ++i) {
    int n = n0 + i;
    if (n >= N_NODES) break;
    int b = batch[n];
    if (b != curb) {
      if (curb >= 0) atomicAdd(&pooled[curb * HC + t], acc);
      acc = 0.f; curb = b;
    }
    acc += outv[(size_t)n * HC + t];
  }
  if (curb >= 0) atomicAdd(&pooled[curb * HC + t], acc);
}

// ---------------- head MLP: concat(means + biases, global) -> relu(W1) -> W2 ----------------
__global__ void k_head(const float* __restrict__ pooled,   // [3][64][128] sums
                       const int* __restrict__ cnt,        // [64]
                       const float* __restrict__ b_ant, const float* __restrict__ b_post,
                       const float* __restrict__ b_lat,
                       const float* __restrict__ gf,       // [64][32]
                       const float* __restrict__ W1, const float* __restrict__ b1,
                       const float* __restrict__ W2, const float* __restrict__ b2,
                       float* __restrict__ out) {          // [64][10]
  __shared__ float hid[NBATCH][CH];
  int t = threadIdx.x;
  for (int p = t; p < NBATCH * CH; p += blockDim.x) {
    int b = p >> 4, j = p & 15;
    float inv = 1.0f / fmaxf((float)cnt[b], 1.0f);
    float s = b1[j];
    for (int c = 0; c < HC; ++c)
      s += (pooled[0 * NBATCH * HC + b * HC + c] * inv + b_ant[c])  * W1[(c) * CH + j];
    for (int c = 0; c < HC; ++c)
      s += (pooled[1 * NBATCH * HC + b * HC + c] * inv + b_post[c]) * W1[(HC + c) * CH + j];
    for (int c = 0; c < HC; ++c)
      s += (pooled[2 * NBATCH * HC + b * HC + c] * inv + b_lat[c])  * W1[(2 * HC + c) * CH + j];
    for (int c = 0; c < GFEAT; ++c)
      s += gf[b * GFEAT + c] * W1[(3 * HC + c) * CH + j];
    hid[b][j] = fmaxf(s, 0.f);
  }
  __syncthreads();
  for (int p = t; p < NBATCH * NCLS; p += blockDim.x) {
    int b = p / NCLS, k = p % NCLS;
    float s = b2[k];
#pragma unroll
    for (int j = 0; j < CH; ++j) s += hid[b][j] * W2[j * NCLS + k];
    out[b * NCLS + k] = s;
  }
}

extern "C" void kernel_launch(void* const* d_in, const int* in_sizes, int n_in,
                              void* d_out, int out_size, void* d_ws, size_t ws_size,
                              hipStream_t stream) {
  (void)in_sizes; (void)n_in; (void)out_size; (void)ws_size;
  // inputs per setup_inputs() dict order
  const float* x_view[3]  = { (const float*)d_in[0], (const float*)d_in[1], (const float*)d_in[2] };
  const int*   ei_view[3] = { (const int*)d_in[3],   (const int*)d_in[4],   (const int*)d_in[5]   };
  const int*   batch      = (const int*)d_in[6];
  const float* gf         = (const float*)d_in[7];
  const float* Wv[3]      = { (const float*)d_in[8],  (const float*)d_in[12], (const float*)d_in[16] };
  const float* asv[3]     = { (const float*)d_in[9],  (const float*)d_in[13], (const float*)d_in[17] };
  const float* adv[3]     = { (const float*)d_in[10], (const float*)d_in[14], (const float*)d_in[18] };
  const float* bv[3]      = { (const float*)d_in[11], (const float*)d_in[15], (const float*)d_in[19] };
  const float* W1 = (const float*)d_in[20];
  const float* b1 = (const float*)d_in[21];
  const float* W2 = (const float*)d_in[22];
  const float* b2 = (const float*)d_in[23];
  float* outp = (float*)d_out;

  // workspace layout (bytes, 256-aligned); total ~70.6 MB, buffers reused per view
  char* ws = (char*)d_ws;
  unsigned short* xbf = (unsigned short*)(ws + 0);            // 12,800,000
  unsigned short* wp  = (unsigned short*)(ws + 12800000);     //     32,768
  float* h      = (float*)(ws + 12832768);                    // 25,600,000
  float* a_src  = (float*)(ws + 38432768);                    //  1,600,000
  float* a_dst  = (float*)(ws + 40032768);                    //  1,600,000
  float* m      = (float*)(ws + 41632768);                    //  1,600,000
  float* den    = (float*)(ws + 43232768);                    //  1,600,000
  float* outv   = (float*)(ws + 44832768);                    // 25,600,000
  float* pooled = (float*)(ws + 70432768);                    //     98,304 (3*64*128)
  int*   cnt    = (int*)  (ws + 70531072);                    //        256

  // zero pooled sums + counts (contiguous region)
  {
    int nWords = (3 * NBATCH * HC) + 64;
    k_zero<<<(nWords + 255) / 256, 256, 0, stream>>>((unsigned int*)pooled, nWords);
  }
  k_count_batch<<<(N_NODES + 255) / 256, 256, 0, stream>>>(batch, cnt);

  for (int v = 0; v < 3; ++v) {
    k_pack_w<<<1, 1024, 0, stream>>>(Wv[v], wp);
    k_cvt_bf16<<<(N_NODES * F_IN / 4 + 255) / 256, 256, 0, stream>>>(x_view[v], xbf, N_NODES * F_IN);
    k_gemm<<<(M_TILES + 7) / 8, 256, 0, stream>>>(xbf, wp, h);
    k_attn_init<<<(N_NODES * HEADS + 255) / 256, 256, 0, stream>>>(
        h, asv[v], adv[v], a_src, a_dst, m, den, outv);
    k_edge_max<<<(EP + 255) / 256, 256, 0, stream>>>(ei_view[v], a_src, a_dst, m);
    k_edge_sum<<<(EP + 255) / 256, 256, 0, stream>>>(ei_view[v], a_src, a_dst, m, den);
    k_edge_aggr<<<(EP + 7) / 8, 256, 0, stream>>>(ei_view[v], a_src, a_dst, m, den, h, outv);
    k_pool<<<(N_NODES + 255) / 256, 128, 0, stream>>>(outv, batch, pooled + v * NBATCH * HC);
  }

  k_head<<<1, 256, 0, stream>>>(pooled, cnt, bv[0], bv[1], bv[2], gf, W1, b1, W2, b2, outp);
}
